// SelfAttention_51702816309663
// MI455X (gfx1250) — matrix operand order repaired and checked by reading, not compiled, not run on hardware
//
#include <hip/hip_runtime.h>
#include <cmath>

// ---------------------------------------------------------------------------
// MI455X (gfx1250) self-attention forward.
// All matmuls via v_wmma_f32_16x16x32_bf16 (wave32, 16x16 tiles, K=32).
// ---------------------------------------------------------------------------

typedef __bf16 bf16_t;
typedef __attribute__((ext_vector_type(16))) __bf16 v16bf;
typedef __attribute__((ext_vector_type(8)))  __bf16 v8bf;
typedef __attribute__((ext_vector_type(4)))  __bf16 v4bf;
typedef __attribute__((ext_vector_type(8)))  float  v8f;

#define H_ 16
#define D_ 64
#define L_ 1024
#define C_ 1024
#define B_ 8
#define BL_ (B_ * L_)
#define MAX_SCALE_MUL 4.605170185988091f  // ln(100)

__device__ __forceinline__ int lane_id() { return (int)(threadIdx.x & 31u); }

// A-fragment (16x32 bf16, MxK). Per ISA table:
//   lanes 0-15 : row M=lane,   elems = K{k0+0..7,  k0+16..23}
//   lanes16-31 : row M=lane-16,elems = K{k0+8..15, k0+24..31}
// `row` is the per-lane row index in a K-contiguous (row-major over K) array.
__device__ __forceinline__ v16bf load_a_frag(const bf16_t* __restrict__ base,
                                             int ld, int row, int k0) {
  const int lane = lane_id();
  const int half8 = (lane < 16) ? 0 : 8;
  const bf16_t* p = base + (size_t)row * ld + k0 + half8;
  v8bf lo = *(const v8bf*)(p);
  v8bf hi = *(const v8bf*)(p + 16);
  v16bf r;
#pragma unroll
  for (int i = 0; i < 8; ++i) { r[i] = lo[i]; r[i + 8] = hi[i]; }
  return r;
}

// B-fragment (32x16 bf16, KxN), column per lane, K-contiguous storage:
//   lanes 0-15 : col N=lane,    elems = K{k0+0..7,  k0+16..23}
//   lanes16-31 : col N=lane-16, elems = K{k0+8..15, k0+24..31}
// `nrow` is the per-lane storage row (= column of B) in a K-contiguous array.
__device__ __forceinline__ v16bf load_b_frag(const bf16_t* __restrict__ base,
                                             int ld, int nrow, int k0) {
  const int lane = lane_id();
  const int k16 = (lane < 16) ? 0 : 8;
  const bf16_t* p = base + (size_t)nrow * ld + k0 + k16;
  v8bf lo = *(const v8bf*)(p);
  v8bf hi = *(const v8bf*)(p + 16);
  v16bf r;
#pragma unroll
  for (int i = 0; i < 8; ++i) { r[i] = lo[i]; r[i + 8] = hi[i]; }
  return r;
}

__device__ __forceinline__ v8f wmma_bf16(v16bf a, v16bf b, v8f c) {
  // D = A*B + C ; args: (neg_a, A, neg_b, B, c_mod, C, reuse_a, reuse_b)
  return __builtin_amdgcn_wmma_f32_16x16x32_bf16(false, a, false, b, (short)0,
                                                 c, false, false);
}

__device__ __forceinline__ float redsum16(float v) {
  v += __shfl_xor(v, 1, 16);
  v += __shfl_xor(v, 2, 16);
  v += __shfl_xor(v, 4, 16);
  v += __shfl_xor(v, 8, 16);
  return v;
}
__device__ __forceinline__ float redmax16(float v) {
  v = fmaxf(v, __shfl_xor(v, 1, 16));
  v = fmaxf(v, __shfl_xor(v, 2, 16));
  v = fmaxf(v, __shfl_xor(v, 4, 16));
  v = fmaxf(v, __shfl_xor(v, 8, 16));
  return v;
}

// ---------------------------------------------------------------------------
// Kernel 0: f32 -> bf16 conversion, 4 elems / thread.
// ---------------------------------------------------------------------------
__global__ __launch_bounds__(256) void cvt_f32_bf16_kernel(
    const float* __restrict__ in, bf16_t* __restrict__ out, int n4) {
  int i = blockIdx.x * blockDim.x + threadIdx.x;
  if (i < n4) {
    float4 v = ((const float4*)in)[i];
    v4bf o;
    o[0] = (bf16_t)v.x; o[1] = (bf16_t)v.y;
    o[2] = (bf16_t)v.z; o[3] = (bf16_t)v.w;
    ((v4bf*)out)[i] = o;
  }
}

// ---------------------------------------------------------------------------
// Kernel 1: qkv = x @ w_qkv.T + bias, fused with l2-norm (q,k), q *= sm[h],
// and scatter into qh[B,H,L,D], kh[B,H,L,D], vTh[B,H,D,L] (bf16).
// Block = 256 threads (8 waves). Tile = 128 rows x 64 cols; wave w owns a
// 16x64 strip (4 WMMA accumulators). Col tile = (sel,h): sel=tile/16, h=tile%16.
// C/D layout: acc[n][r] lives at (row = r + 8*(lane>=16), col = n*16 + lane%16).
// ---------------------------------------------------------------------------
__global__ __launch_bounds__(256) void qkv_gemm_norm_kernel(
    const bf16_t* __restrict__ xh, const bf16_t* __restrict__ wh,
    const float* __restrict__ q_bias, const float* __restrict__ v_bias,
    const float* __restrict__ scale_mul, bf16_t* __restrict__ qh,
    bf16_t* __restrict__ kh, bf16_t* __restrict__ vTh) {
  const int lane = lane_id();
  const int wv = threadIdx.x >> 5;
  const int ln16 = lane & 15;
  const int hi8 = (lane >> 4) << 3;
  const int rowBase = blockIdx.x * 128 + wv * 16;
  const int colTile = blockIdx.y;  // 0..47
  const int sel = colTile >> 4;    // 0=q 1=k 2=v
  const int h = colTile & 15;
  const int nBase = colTile * 64;
  const int arow = rowBase + ln16;

  v8f acc[4];
#pragma unroll
  for (int n = 0; n < 4; ++n)
#pragma unroll
    for (int i = 0; i < 8; ++i) acc[n][i] = 0.0f;

  for (int kk = 0; kk < C_; kk += 32) {
    v16bf va = load_a_frag(xh, C_, arow, kk);
#pragma unroll
    for (int n = 0; n < 4; ++n) {
      v16bf vb = load_b_frag(wh, C_, nBase + n * 16 + ln16, kk);
      acc[n] = wmma_bf16(va, vb, acc[n]);
    }
  }

  // bias (column-only): q_bias for sel==0, zeros for k, v_bias for sel==2
#pragma unroll
  for (int n = 0; n < 4; ++n) {
    const int hd = h * 64 + n * 16 + ln16;  // index within C
    float bv = (sel == 0) ? q_bias[hd] : ((sel == 2) ? v_bias[hd] : 0.0f);
#pragma unroll
    for (int r = 0; r < 8; ++r) acc[n][r] += bv;
  }

  if (sel < 2) {
    const float mul =
        (sel == 0) ? __expf(fminf(scale_mul[h], MAX_SCALE_MUL)) : 1.0f;
#pragma unroll
    for (int r = 0; r < 8; ++r) {
      float ss = 0.0f;
#pragma unroll
      for (int n = 0; n < 4; ++n) { float t = acc[n][r]; ss += t * t; }
      ss = redsum16(ss);  // 64-col row sum across the 16-lane half
      float rn = mul / fmaxf(sqrtf(ss), 1e-12f);
#pragma unroll
      for (int n = 0; n < 4; ++n) acc[n][r] *= rn;
    }
  }

#pragma unroll
  for (int r = 0; r < 8; ++r) {
    const int row = rowBase + r + hi8;  // global row in [0, B*L)
    const int b = row >> 10;
    const int l = row & 1023;
#pragma unroll
    for (int n = 0; n < 4; ++n) {
      const int d = n * 16 + ln16;  // 0..63 within head
      const bf16_t v = (bf16_t)acc[n][r];
      const size_t bh = (size_t)(b * H_ + h);
      if (sel == 0)
        qh[(((bh << 10) | l) << 6) + d] = v;
      else if (sel == 1)
        kh[(((bh << 10) | l) << 6) + d] = v;
      else
        vTh[(((bh << 6) | d) << 10) + l] = v;  // transposed: [B,H,D,L]
    }
  }
}

// ---------------------------------------------------------------------------
// Kernel 2: flash attention for one (b,h). Block = 128 threads (4 waves),
// each wave owns 16 query rows. Key loop in chunks of 32; online softmax with
// per-lane row stats (rows r..r live in VGPR index r; lanes<16 rows 0-7,
// lanes>=16 rows 8-15; width-16 shuffles reduce across the 32 columns).
// P (C-layout) is staged through LDS to A-layout for the P@V WMMA.
// ---------------------------------------------------------------------------
__global__ __launch_bounds__(128) void attn_kernel(
    const bf16_t* __restrict__ qh, const bf16_t* __restrict__ kh,
    const bf16_t* __restrict__ vTh, const float* __restrict__ attn_bias,
    bf16_t* __restrict__ oh) {
  __shared__ bf16_t ldsP[4][16][32];  // per-wave 16x32 P tile (1 KB/wave)
  const int lane = lane_id();
  const int wv = threadIdx.x >> 5;
  const int ln16 = lane & 15;
  const int hi8 = (lane >> 4) << 3;
  const int bh = blockIdx.y;  // b*H + h
  const int b = bh >> 4;
  const int h = bh & 15;
  const int qbase = blockIdx.x * 64 + wv * 16;

  const bf16_t* qp = qh + ((size_t)bh << 16);   // [L,D]
  const bf16_t* kp = kh + ((size_t)bh << 16);   // [L,D]
  const bf16_t* vp = vTh + ((size_t)bh << 16);  // [D,L]
  const float* bp = attn_bias + ((size_t)h << 20);

  // Q A-fragments kept resident: K = d in two 32-chunks
  v16bf aq[2];
#pragma unroll
  for (int kb = 0; kb < 2; ++kb)
    aq[kb] = load_a_frag(qp, D_, qbase + ln16, kb * 32);

  float m[8], lsum[8];
#pragma unroll
  for (int r = 0; r < 8; ++r) { m[r] = -INFINITY; lsum[r] = 0.0f; }
  v8f accO[4];
#pragma unroll
  for (int n = 0; n < 4; ++n)
#pragma unroll
    for (int i = 0; i < 8; ++i) accO[n][i] = 0.0f;

  for (int jb = 0; jb < L_; jb += 32) {
    // S strip = q_tile @ k_tile^T   (16 x 32)
    v8f s[2];
#pragma unroll
    for (int n2 = 0; n2 < 2; ++n2) {
#pragma unroll
      for (int i = 0; i < 8; ++i) s[n2][i] = 0.0f;
#pragma unroll
      for (int kb = 0; kb < 2; ++kb) {
        v16bf vb = load_b_frag(kp, D_, jb + n2 * 16 + ln16, kb * 32);
        s[n2] = wmma_bf16(aq[kb], vb, s[n2]);
      }
    }
    // bias + online softmax per row, write P tile to LDS as bf16
#pragma unroll
    for (int r = 0; r < 8; ++r) {
      const int lq = qbase + r + hi8;
      const float* brow = bp + ((size_t)lq << 10) + jb + ln16;
      float s0 = s[0][r] + brow[0];
      float s1 = s[1][r] + brow[16];
      float mx = redmax16(fmaxf(s0, s1));
      float mnew = fmaxf(m[r], mx);
      float sc = __expf(m[r] - mnew);  // 0 on first iteration (m = -inf)
      float p0 = __expf(s0 - mnew);
      float p1 = __expf(s1 - mnew);
      float rs = redsum16(p0 + p1);
      lsum[r] = lsum[r] * sc + rs;
      m[r] = mnew;
#pragma unroll
      for (int n = 0; n < 4; ++n) accO[n][r] *= sc;
      ldsP[wv][r + hi8][ln16] = (bf16_t)p0;
      ldsP[wv][r + hi8][16 + ln16] = (bf16_t)p1;
    }
    __syncthreads();
    // P@V: A = P (16x32, K = key index), B = vT rows (K-contiguous)
    v16bf pa = load_a_frag(&ldsP[wv][0][0], 32, ln16, 0);
#pragma unroll
    for (int n = 0; n < 4; ++n) {
      v16bf vb = load_b_frag(vp, L_, n * 16 + ln16, jb);
      accO[n] = wmma_bf16(pa, vb, accO[n]);
    }
    __syncthreads();
  }

  // normalize by softmax denominator, write [B, L, h*64+d] bf16
#pragma unroll
  for (int r = 0; r < 8; ++r) {
    const float inv = 1.0f / lsum[r];
    const int lq = qbase + r + hi8;
    const size_t obase = (((size_t)(b * L_ + lq)) << 10) + h * 64;
#pragma unroll
    for (int n = 0; n < 4; ++n)
      oh[obase + n * 16 + ln16] = (bf16_t)(accO[n][r] * inv);
  }
}

// ---------------------------------------------------------------------------
// Kernel 3: out = attn_out @ w_proj.T + b_proj  (f32 output)
// ---------------------------------------------------------------------------
__global__ __launch_bounds__(256) void proj_gemm_kernel(
    const bf16_t* __restrict__ oh, const bf16_t* __restrict__ wh,
    const float* __restrict__ b_proj, float* __restrict__ out) {
  const int lane = lane_id();
  const int wv = threadIdx.x >> 5;
  const int ln16 = lane & 15;
  const int hi8 = (lane >> 4) << 3;
  const int rowBase = blockIdx.x * 128 + wv * 16;
  const int nBase = blockIdx.y * 64;

  v8f acc[4];
#pragma unroll
  for (int n = 0; n < 4; ++n)
#pragma unroll
    for (int i = 0; i < 8; ++i) acc[n][i] = 0.0f;

  for (int kk = 0; kk < C_; kk += 32) {
    v16bf va = load_a_frag(oh, C_, rowBase + ln16, kk);
#pragma unroll
    for (int n = 0; n < 4; ++n) {
      v16bf vb = load_b_frag(wh, C_, nBase + n * 16 + ln16, kk);
      acc[n] = wmma_bf16(va, vb, acc[n]);
    }
  }
#pragma unroll
  for (int r = 0; r < 8; ++r) {
    const int row = rowBase + r + hi8;
#pragma unroll
    for (int n = 0; n < 4; ++n) {
      const int col = nBase + n * 16 + ln16;
      out[((size_t)row << 10) + col] = acc[n][r] + b_proj[col];
    }
  }
}

// ---------------------------------------------------------------------------
extern "C" void kernel_launch(void* const* d_in, const int* in_sizes, int n_in,
                              void* d_out, int out_size, void* d_ws,
                              size_t ws_size, hipStream_t stream) {
  const float* x = (const float*)d_in[0];
  const float* attn_bias = (const float*)d_in[1];
  const float* w_qkv = (const float*)d_in[2];
  const float* q_bias = (const float*)d_in[3];
  const float* v_bias = (const float*)d_in[4];
  const float* scale_mul = (const float*)d_in[5];
  const float* w_proj = (const float*)d_in[6];
  const float* b_proj = (const float*)d_in[7];
  float* out = (float*)d_out;

  // workspace layout (bf16), 88 MB total
  char* ws = (char*)d_ws;
  const size_t MB = (size_t)1 << 20;
  bf16_t* xh = (bf16_t*)(ws);             // [BL, C]      16 MB
  bf16_t* wqh = (bf16_t*)(ws + 16 * MB);  // [3C, C]       6 MB
  bf16_t* wph = (bf16_t*)(ws + 22 * MB);  // [C, C]        2 MB
  bf16_t* qh = (bf16_t*)(ws + 24 * MB);   // [B,H,L,D]    16 MB
  bf16_t* kh = (bf16_t*)(ws + 40 * MB);   // [B,H,L,D]    16 MB
  bf16_t* vTh = (bf16_t*)(ws + 56 * MB);  // [B,H,D,L]    16 MB
  bf16_t* oh = (bf16_t*)(ws + 72 * MB);   // [BL, C]      16 MB

  int n4 = (BL_ * C_) / 4;
  cvt_f32_bf16_kernel<<<(n4 + 255) / 256, 256, 0, stream>>>(x, xh, n4);
  n4 = (3 * C_ * C_) / 4;
  cvt_f32_bf16_kernel<<<(n4 + 255) / 256, 256, 0, stream>>>(w_qkv, wqh, n4);
  n4 = (C_ * C_) / 4;
  cvt_f32_bf16_kernel<<<(n4 + 255) / 256, 256, 0, stream>>>(w_proj, wph, n4);

  qkv_gemm_norm_kernel<<<dim3(BL_ / 128, 48), 256, 0, stream>>>(
      xh, wqh, q_bias, v_bias, scale_mul, qh, kh, vTh);

  attn_kernel<<<dim3(L_ / 64, B_ * H_), 128, 0, stream>>>(qh, kh, vTh,
                                                          attn_bias, oh);

  proj_gemm_kernel<<<dim3(BL_ / 128, C_ / 64), 256, 0, stream>>>(oh, wph,
                                                                 b_proj, out);
  (void)in_sizes; (void)n_in; (void)out_size; (void)ws_size;
}
